// TriangleMultiplication_45664092291507
// MI455X (gfx1250) — matrix operand order repaired and checked by reading, not compiled, not run on hardware
//
#include <hip/hip_runtime.h>
#include <cstdint>
#include <cstddef>

// ---------------------------------------------------------------------------
// Triangle multiplication (AlphaFold-style), MI455X / gfx1250.
//
// Roofline: einsum = 128 GEMMs of 512x512x512 = 34.4 TFLOP  (compute bound,
//           ~14 ms at bf16-WMMA-class rates); total HBM traffic ~0.5 GB =
//           ~20 us @ 23.3 TB/s (negligible). => einsum on
//           v_wmma_f32_16x16x32_bf16, global->LDS async double buffering.
// ---------------------------------------------------------------------------

constexpr int NN = 512;   // sequence length
constexpr int DD = 128;   // pair channel dim
constexpr float LN_EPS = 1e-5f;

typedef __attribute__((ext_vector_type(16))) __bf16 v16bf;
typedef __attribute__((ext_vector_type(8)))  float  v8f;

union FragAB { v16bf v; uint4 u[2]; };

__device__ __forceinline__ unsigned short f2bf(float f) {
  unsigned int u = __float_as_uint(f);
  u += 0x7fffu + ((u >> 16) & 1u);          // round-to-nearest-even
  return (unsigned short)(u >> 16);
}
__device__ __forceinline__ float bf2f(unsigned short s) {
  return __uint_as_float(((unsigned int)s) << 16);
}
__device__ __forceinline__ float sigmoidf_(float x) {
  return 1.0f / (1.0f + __expf(-x));
}

// ---------------------------------------------------------------------------
// Kernel 1: LayerNorm(pair) -> xln (bf16, [i][k][d]); gate/proj matvecs;
//           a,b (bf16) written D-MAJOR: a_bf[d][i][k], b_bf[d][i][k] so the
//           einsum becomes 128 contiguous row-major NT GEMMs.
// Block: 128 threads, row i, 128 consecutive k. LDS row pad 129
// (bank = (pos + c) % 64 -> conflict-free per-thread row reads).
// Output-channel loop tiled by 8: 1 LDS read feeds 16 FMAs (dual-issuable).
// ---------------------------------------------------------------------------
constexpr int APAD = 129;

__global__ void prep_kernel(const float* __restrict__ pair,
                            const float* __restrict__ mask,
                            const float* __restrict__ pw, const float* __restrict__ pb,
                            const float* __restrict__ gw, const float* __restrict__ gb,
                            unsigned short* __restrict__ a_out,
                            unsigned short* __restrict__ b_out,
                            unsigned short* __restrict__ xln_out) {
  extern __shared__ float x[];                  // [128][APAD]
  const int t  = threadIdx.x;                   // 128 threads
  const int i  = blockIdx.x >> 2;
  const int k0 = (blockIdx.x & 3) * 128;

  const float* src = pair + ((size_t)i * NN + k0) * DD;
  for (int idx = t; idx < 128 * DD; idx += 128)
    x[(idx >> 7) * APAD + (idx & (DD - 1))] = src[idx];
  __syncthreads();

  float* xr = x + t * APAD;                     // this thread's position
  float mu = 0.f;
  #pragma unroll 8
  for (int c = 0; c < DD; ++c) mu += xr[c];
  mu *= (1.0f / DD);
  float var = 0.f;
  #pragma unroll 8
  for (int c = 0; c < DD; ++c) { float dv = xr[c] - mu; var += dv * dv; }
  var *= (1.0f / DD);
  const float rs = rsqrtf(var + LN_EPS);
  #pragma unroll 8
  for (int c = 0; c < DD; ++c) xr[c] = (xr[c] - mu) * rs;
  __syncthreads();

  // coalesced bf16 write of LN(pair) in natural [i][k][d] layout
  unsigned short* xd = xln_out + ((size_t)i * NN + k0) * DD;
  for (int idx = t; idx < 128 * DD; idx += 128)
    xd[idx] = f2bf(x[(idx >> 7) * APAD + (idx & (DD - 1))]);

  // gate/proj: 256 outputs in tiles of 8; weights wave-uniform (scalar loads)
  const float  m       = mask[(size_t)i * NN + k0 + t];
  const size_t colbase = (size_t)i * NN + k0 + t;
  for (int ot = 0; ot < 2 * DD; ot += 8) {
    float ag[8], ap[8];
    #pragma unroll
    for (int j = 0; j < 8; ++j) { ag[j] = gb[ot + j]; ap[j] = pb[ot + j]; }
    for (int c = 0; c < DD; ++c) {
      const float xv = xr[c];
      #pragma unroll
      for (int j = 0; j < 8; ++j) {
        ag[j] = fmaf(xv, gw[(ot + j) * DD + c], ag[j]);
        ap[j] = fmaf(xv, pw[(ot + j) * DD + c], ap[j]);
      }
    }
    #pragma unroll
    for (int j = 0; j < 8; ++j) {
      const int o = ot + j;
      unsigned short bv = f2bf(m * sigmoidf_(ag[j]) * ap[j]);
      // fixed o, consecutive lanes -> consecutive k: coalesced b16 stores
      if (o < DD) a_out[(size_t)o        * (NN * NN) + colbase] = bv;
      else        b_out[(size_t)(o - DD) * (NN * NN) + colbase] = bv;
    }
  }
}

// ---------------------------------------------------------------------------
// Kernel 2: tri[d] = A_d @ B_d^T with v_wmma_f32_16x16x32_bf16.
// Block: 256 thr (8 waves), 128x128 C tile, K-step 32, wave tile 64x32
// (8 v8f accumulators). __launch_bounds__(256,1) lifts the occupancy-driven
// 128-VGPR cap so accumulators never spill.
// Global->LDS via GLOBAL_LOAD_ASYNC_TO_LDS_B128 (ASYNCcnt), double-buffered
// (two barriers/stage keeps the scheduler's
// regions small => low register pressure):
//   issue stage k+1  -> s_wait_asynccnt 4 (stage k landed; loads retire in
//   order) -> barrier -> frags + 8 WMMA -> barrier (reads drained before the
//   other buffer is overwritten next iteration).
// LDS rows padded to 56 bf16 (112 B: 16B-aligned b128; 28-bank row stride ->
// 16 consecutive rows hit 16 distinct banks). Branchless chunk issue.
// Fragment gathers follow ISA 7.12.2 16-bit A/B layouts.
// ---------------------------------------------------------------------------
constexpr int BK  = 32;
constexpr int LDP = 56;
constexpr int STG = 2 * 128 * LDP;      // ushorts per stage (A tile + B tile)

__device__ __forceinline__ void issue_tile_async(
    const unsigned short* __restrict__ Ad, const unsigned short* __restrict__ Bd,
    unsigned short* __restrict__ As, unsigned short* __restrict__ Bs,
    int i0, int j0, int k0, int t) {
  // 128 rows x 64 B per operand; 512 16-B chunks over 256 threads, no guards
  #pragma unroll
  for (int s = 0; s < 2; ++s) {
    const int cid = t + s * 256;
    const int r = cid >> 2, q = cid & 3;
    const unsigned long long ga =
        (unsigned long long)(uintptr_t)(Ad + (size_t)(i0 + r) * NN + k0 + q * 8);
    const unsigned long long gb =
        (unsigned long long)(uintptr_t)(Bd + (size_t)(j0 + r) * NN + k0 + q * 8);
    // flat shared pointer keeps the LDS byte offset in addr[31:0] (ISA 10.2)
    const unsigned la = (unsigned)(uintptr_t)(As + r * LDP + q * 8);
    const unsigned lb = (unsigned)(uintptr_t)(Bs + r * LDP + q * 8);
    asm volatile("global_load_async_to_lds_b128 %0, %1, off"
                 :: "v"(la), "v"(ga) : "memory");
    asm volatile("global_load_async_to_lds_b128 %0, %1, off"
                 :: "v"(lb), "v"(gb) : "memory");
  }
}

__global__ void __launch_bounds__(256, 1)
tri_kernel(const unsigned short* __restrict__ Abf,
           const unsigned short* __restrict__ Bbf,
           float* __restrict__ Tri) {
  extern __shared__ unsigned short sm[];        // 2 stages x [A[128][LDP], B[128][LDP]]

  const int d  = blockIdx.z;
  const int i0 = blockIdx.y * 128;
  const int j0 = blockIdx.x * 128;
  const unsigned short* Ad = Abf + (size_t)d * NN * NN;
  const unsigned short* Bd = Bbf + (size_t)d * NN * NN;
  float* Td = Tri + (size_t)d * NN * NN;

  const int t    = threadIdx.x;
  const int lane = t & 31;
  const int w    = t >> 5;
  const int wm   = w & 1;     // 2 waves over M (64 rows each)
  const int wn   = w >> 1;    // 4 waves over N (32 cols each)
  const int ml   = lane & 15;
  const int h    = lane >> 4;

  v8f acc[4][2] = {};

  // prologue: stage 0
  issue_tile_async(Ad, Bd, sm, sm + 128 * LDP, i0, j0, 0, t);

  constexpr int KT = NN / BK;                   // 16 stages
  for (int kt = 0; kt < KT; ++kt) {
    unsigned short* As = sm + (kt & 1) * STG;
    unsigned short* Bs = As + 128 * LDP;

    if (kt + 1 < KT) {
      unsigned short* An = sm + ((kt + 1) & 1) * STG;
      issue_tile_async(Ad, Bd, An, An + 128 * LDP, i0, j0, (kt + 1) * BK, t);
      asm volatile("s_wait_asynccnt 0x4" ::: "memory");  // stage kt landed
    } else {
      asm volatile("s_wait_asynccnt 0x0" ::: "memory");
    }
    __syncthreads();                             // stage kt globally complete

    FragAB af[4], bfr[2];
    #pragma unroll
    for (int f = 0; f < 4; ++f) {   // A: lane(m=ml,h): K {8h..8h+7},{16+8h..}
      const unsigned short* p = As + (wm * 64 + f * 16 + ml) * LDP + h * 8;
      af[f].u[0] = *(const uint4*)(p);
      af[f].u[1] = *(const uint4*)(p + 16);
    }
    #pragma unroll
    for (int f = 0; f < 2; ++f) {   // B: lane(n=ml,h): K {8h..8h+7},{16+8h..}
      const unsigned short* p = Bs + (wn * 32 + f * 16 + ml) * LDP + h * 8;
      bfr[f].u[0] = *(const uint4*)(p);
      bfr[f].u[1] = *(const uint4*)(p + 16);
    }

    #pragma unroll
    for (int a = 0; a < 4; ++a)
      #pragma unroll
      for (int b = 0; b < 2; ++b)
        acc[a][b] = __builtin_amdgcn_wmma_f32_16x16x32_bf16(
            false, af[a].v, false, bfr[b].v, (short)0, acc[a][b], false, false);

    __syncthreads();                 // everyone done reading before overwrite
  }

  // C layout: elem r of v8f -> M = r + 8*h, N = ml  => 64 B coalesced runs
  #pragma unroll
  for (int a = 0; a < 4; ++a) {
    const int Mb = i0 + wm * 64 + a * 16 + h * 8;
    #pragma unroll
    for (int b = 0; b < 2; ++b) {
      const int col = j0 + wn * 32 + b * 16 + ml;
      #pragma unroll
      for (int r = 0; r < 8; ++r)
        Td[(size_t)(Mb + r) * NN + col] = acc[a][b][r];
    }
  }
}

// ---------------------------------------------------------------------------
// Kernel 3: out = sigmoid(xln @ ogw^T + ogb) * (LN(tri) @ opw^T + opb).
// Block: 128 thr over a 4(i) x 16(j) = 64-position tile; thread (pos, half)
// computes 64 output channels in tiles of 8 (2 LDS reads -> 16 FMAs).
// tri slab staged [d][pos] (bank = pos, conflict-free); xln & out staged with
// row pad 129. Output buffered in LDS then written coalesced as [i][j][d].
// ---------------------------------------------------------------------------
constexpr int XP = 129;

__global__ void out_kernel(const unsigned short* __restrict__ xln,
                           const float* __restrict__ tri,
                           const float* __restrict__ opw, const float* __restrict__ opb,
                           const float* __restrict__ ogw, const float* __restrict__ ogb,
                           float* __restrict__ out) {
  extern __shared__ float smf[];
  float* triS = smf;                  // [128][64]
  float* xS   = smf + 128 * 64;       // [64][XP]
  float* oS   = xS + 64 * XP;         // [64][XP]

  const int t  = threadIdx.x;         // 128
  const int i0 = blockIdx.y * 4;
  const int j0 = blockIdx.x * 16;

  for (int idx = t; idx < 128 * 64; idx += 128) {
    int dd = idx >> 6, p = idx & 63;
    triS[idx] = tri[((size_t)dd * NN + i0 + (p >> 4)) * NN + j0 + (p & 15)];
  }
  for (int idx = t; idx < 64 * DD; idx += 128) {
    int p = idx >> 7, c = idx & 127;
    xS[p * XP + c] =
        bf2f(xln[((size_t)(i0 + (p >> 4)) * NN + j0 + (p & 15)) * DD + c]);
  }
  __syncthreads();

  const int pos = t & 63;
  const int oh  = t >> 6;             // output-channel half

  float mu = 0.f;
  #pragma unroll 8
  for (int dd = 0; dd < DD; ++dd) mu += triS[dd * 64 + pos];
  mu *= (1.0f / DD);
  float var = 0.f;
  #pragma unroll 8
  for (int dd = 0; dd < DD; ++dd) { float dv = triS[dd * 64 + pos] - mu; var += dv * dv; }
  var *= (1.0f / DD);
  const float rs = rsqrtf(var + LN_EPS);

  const float* xrow = xS + pos * XP;
  for (int ot = oh * 64; ot < oh * 64 + 64; ot += 8) {
    float g[8], pv[8];
    #pragma unroll
    for (int j = 0; j < 8; ++j) { g[j] = ogb[ot + j]; pv[j] = opb[ot + j]; }
    for (int c = 0; c < DD; ++c) {
      const float xv = xrow[c];
      const float tv = (triS[c * 64 + pos] - mu) * rs;
      #pragma unroll
      for (int j = 0; j < 8; ++j) {
        g[j]  = fmaf(xv, ogw[(ot + j) * DD + c], g[j]);
        pv[j] = fmaf(tv, opw[(ot + j) * DD + c], pv[j]);
      }
    }
    #pragma unroll
    for (int j = 0; j < 8; ++j)
      oS[pos * XP + ot + j] = sigmoidf_(g[j]) * pv[j];
  }
  __syncthreads();

  for (int idx = t; idx < 64 * DD; idx += 128) {
    int p = idx >> 7, c = idx & 127;
    out[((size_t)(i0 + (p >> 4)) * NN + j0 + (p & 15)) * DD + c] = oS[p * XP + c];
  }
}

// ---------------------------------------------------------------------------
// Launch. Workspace layout (320 MiB total):
//   [  0..64M )  xln  bf16 [i][k][d]
//   [ 64..128M)  a_bf bf16 [d][i][k]
//   [128..192M)  b_bf bf16 [d][j][k]
//   [192..320M)  tri  f32  [d][i][j]
// ---------------------------------------------------------------------------
extern "C" void kernel_launch(void* const* d_in, const int* in_sizes, int n_in,
                              void* d_out, int out_size, void* d_ws, size_t ws_size,
                              hipStream_t stream) {
  (void)in_sizes; (void)n_in; (void)out_size; (void)ws_size;
  const float* pair = (const float*)d_in[0];
  const float* mask = (const float*)d_in[1];
  const float* pw   = (const float*)d_in[2];   // ab_proj_w (256,128)
  const float* pb   = (const float*)d_in[3];
  const float* gw   = (const float*)d_in[4];   // ab_gate_w (256,128)
  const float* gb   = (const float*)d_in[5];
  const float* opw  = (const float*)d_in[6];   // out_proj_w (128,128)
  const float* opb  = (const float*)d_in[7];
  const float* ogw  = (const float*)d_in[8];   // out_gate_w (128,128)
  const float* ogb  = (const float*)d_in[9];
  float* out = (float*)d_out;

  char* ws = (char*)d_ws;
  unsigned short* xln  = (unsigned short*)(ws);
  unsigned short* a_bf = (unsigned short*)(ws + ((size_t)64  << 20));
  unsigned short* b_bf = (unsigned short*)(ws + ((size_t)128 << 20));
  float*          tri  = (float*)         (ws + ((size_t)192 << 20));

  prep_kernel<<<NN * 4, 128, 128 * APAD * sizeof(float), stream>>>(
      pair, mask, pw, pb, gw, gb, a_bf, b_bf, xln);

  dim3 gB(4, 4, 128);
  tri_kernel<<<gB, 256, 2 * STG * sizeof(unsigned short), stream>>>(
      a_bf, b_bf, tri);

  dim3 gC(NN / 16, NN / 4);
  out_kernel<<<gC, 128, (128 * 64 + 2 * 64 * XP) * sizeof(float), stream>>>(
      xln, tri, opw, opb, ogw, ogb, out);
}
